// InverseTransitionModel_43224550867249
// MI455X (gfx1250) — compile-verified, matrix operand order checked by reading, not run on hardware
//
#include <hip/hip_runtime.h>
#include <hip/hip_bf16.h>

typedef __bf16 bf16;
typedef __attribute__((ext_vector_type(16))) __bf16 v16bf;
typedef __attribute__((ext_vector_type(8)))  float  v8f;

#define WMMA_BF16(a, b, c) \
  __builtin_amdgcn_wmma_f32_16x16x32_bf16(false, (a), false, (b), (short)0, (c), false, false)

__device__ __forceinline__ bf16  f2b(float f) { return (bf16)f; }   // RNE hw cvt
__device__ __forceinline__ float b2f(bf16 b)  { return (float)b; }

union Frag { v16bf v; unsigned u[8]; };

// Load one 16x32 bf16 fragment (A from activations [M][K] row-major, or
// B from weights stored [N][K] row-major, i.e. B = W^T tile).
// Per-lane layout (ISA 7.12.2, 16-bit A 16x32): lane&15 = row/col index,
// hi = lane>>4 selects K+8; VGPR v holds K = (v&3)*2 + hi*8 + (v>>2)*16.
// The two 16-byte groups vectorize into two b128 loads.
__device__ __forceinline__ v16bf load_frag(const bf16* __restrict__ base, int ld,
                                           int rc0, int k0, int lane) {
  const bf16* p = base + (rc0 + (lane & 15)) * ld + k0 + ((lane >> 4) & 1) * 8;
  Frag f;
  f.u[0] = *(const unsigned*)(p + 0);
  f.u[1] = *(const unsigned*)(p + 2);
  f.u[2] = *(const unsigned*)(p + 4);
  f.u[3] = *(const unsigned*)(p + 6);
  f.u[4] = *(const unsigned*)(p + 16);
  f.u[5] = *(const unsigned*)(p + 18);
  f.u[6] = *(const unsigned*)(p + 20);
  f.u[7] = *(const unsigned*)(p + 22);
  return f.v;
}

// One 16x16 output tile: acc = sum_k A[m0:,k] * B(W^T)[k, n0:]
__device__ __forceinline__ v8f gemm_tile(const bf16* __restrict__ A, int lda, int m0,
                                         const bf16* __restrict__ W, int ldw, int n0,
                                         int K, int lane) {
  v8f acc;
  for (int i = 0; i < 8; ++i) acc[i] = 0.f;
  for (int k0 = 0; k0 < K; k0 += 32) {
    v16bf a = load_frag(A, lda, m0, k0, lane);
    v16bf b = load_frag(W, ldw, n0, k0, lane);
    acc = WMMA_BF16(a, b, acc);
  }
  return acc;
}

// ---------------------------------------------------------------------------
// Weight packing kernels (fp32 -> bf16, once per launch; ~2MB, L2-resident)
// ---------------------------------------------------------------------------
__global__ void cvt_kernel(const float* __restrict__ s, bf16* __restrict__ d, int n) {
  int i = blockIdx.x * 256 + threadIdx.x;
  if (i < n) d[i] = f2b(s[i]);
}

// dst[r][k] (ldk wide): k<kin -> Wih[r][k]; kin<=k<kin+kh -> Whh[r][k-kin]; else 0
__global__ void pack_cat_kernel(const float* __restrict__ Wih, const float* __restrict__ Whh,
                                bf16* __restrict__ dst, int kin, int kh, int ldk, int n) {
  int i = blockIdx.x * 256 + threadIdx.x;
  if (i >= n) return;
  int r = i / ldk, k = i - r * ldk;
  float v = 0.f;
  if (k < kin) v = Wih[r * kin + k];
  else if (k < kin + kh) v = Whh[r * kh + (k - kin)];
  dst[i] = f2b(v);
}

// dst = bf16(Whh rows 512..767)   (the "n"-gate hidden weights)
__global__ void pack_hn_kernel(const float* __restrict__ Whh, bf16* __restrict__ d) {
  int i = blockIdx.x * 256 + threadIdx.x;
  if (i < 256 * 256) d[i] = f2b(Whh[512 * 256 + i]);
}

// Wa [8][256] -> bf16 [16][256] zero-padded panel for one WMMA tile
__global__ void pack_wa_kernel(const float* __restrict__ Wa, bf16* __restrict__ d) {
  int i = blockIdx.x * 256 + threadIdx.x;
  if (i < 16 * 256) {
    int r = i >> 8, k = i & 255;
    d[i] = f2b(r < 8 ? Wa[r * 256 + k] : 0.f);
  }
}

// bS[i] = bih[i]+bhh[i] (768); bN[i-512] = bhh[i] for i>=512
__global__ void bias_kernel(const float* __restrict__ bih, const float* __restrict__ bhh,
                            float* __restrict__ bS, float* __restrict__ bN) {
  int i = blockIdx.x * 256 + threadIdx.x;
  if (i < 768) {
    bS[i] = bih[i] + bhh[i];
    if (i >= 512) bN[i - 512] = bhh[i];
  }
}

// ---------------------------------------------------------------------------
// Encoder: h = relu(relu(LN(X@W1^T+b1))@W2^T+b2); Hinit = h@Winit^T+binit
// 32 batch rows per workgroup, 8 waves. X = [init|final] converted on the fly,
// staged 128 K-columns per barrier round.
// ---------------------------------------------------------------------------
__global__ __launch_bounds__(256) void encoder_kernel(
    const float* __restrict__ ini, const float* __restrict__ fin,
    const bf16* __restrict__ W1bf, const float* __restrict__ b1,
    const float* __restrict__ lng, const float* __restrict__ lnb,
    const bf16* __restrict__ W2bf, const float* __restrict__ b2,
    const bf16* __restrict__ Wibf, const float* __restrict__ binit,
    float* __restrict__ Hinit) {
  __shared__ bf16  xs[32][136];      // 128-wide K staging of X (bf16)
  __shared__ float hbuf[32][264];    // fp32 pre-LN activations
  __shared__ bf16  hbf[32][264];     // bf16 activations (A for GEMM2/3)

  const int t = threadIdx.x, lane = t & 31, w = t >> 5;
  const int b0 = blockIdx.x * 32;
  const int m0 = (w & 1) * 16;
  const int hi = (lane >> 4) & 1, nl = lane & 15;
  const int nb = (w >> 1);

  // ---- GEMM1: [32,1024] x [1024,256]
  v8f acc[4];
  for (int i = 0; i < 4; ++i) for (int r = 0; r < 8; ++r) acc[i][r] = 0.f;
  const int srow = t >> 3, sc = (t & 7) * 16;   // 16 contiguous floats per thread
  for (int k0 = 0; k0 < 1024; k0 += 128) {
    const float* src = (k0 < 512)
        ? (ini + (long)(b0 + srow) * 512 + k0 + sc)
        : (fin + (long)(b0 + srow) * 512 + (k0 - 512) + sc);
    for (int j = 0; j < 16; ++j) xs[srow][sc + j] = f2b(src[j]);
    __syncthreads();
    for (int kk = 0; kk < 128; kk += 32) {
      v16bf a = load_frag(&xs[0][0], 136, m0, kk, lane);
      for (int i = 0; i < 4; ++i) {
        int n0 = (nb * 4 + i) * 16;
        v16bf b = load_frag(W1bf, 1024, n0, k0 + kk, lane);
        acc[i] = WMMA_BF16(a, b, acc[i]);
      }
    }
    __syncthreads();
  }
  for (int i = 0; i < 4; ++i) {
    int n0 = (nb * 4 + i) * 16;
    for (int r = 0; r < 8; ++r)
      hbuf[m0 + r + hi * 8][n0 + nl] = acc[i][r] + b1[n0 + nl];
  }
  __syncthreads();

  // ---- LayerNorm + ReLU (one thread per row)
  if (t < 32) {
    float s = 0.f, s2 = 0.f;
    for (int j = 0; j < 256; ++j) { float v = hbuf[t][j]; s += v; s2 += v * v; }
    float mu = s * (1.f / 256.f);
    float var = s2 * (1.f / 256.f) - mu * mu;
    float rstd = rsqrtf(var + 1e-5f);
    for (int j = 0; j < 256; ++j) {
      float v = (hbuf[t][j] - mu) * rstd * lng[j] + lnb[j];
      hbf[t][j] = f2b(fmaxf(v, 0.f));
    }
  }
  __syncthreads();

  // ---- GEMM2: [32,256] x [256,256] + ReLU, result back into hbf
  for (int i = 0; i < 4; ++i) {
    int n0 = (nb * 4 + i) * 16;
    acc[i] = gemm_tile(&hbf[0][0], 264, m0, W2bf, 256, n0, 256, lane);
  }
  __syncthreads();
  for (int i = 0; i < 4; ++i) {
    int n0 = (nb * 4 + i) * 16;
    for (int r = 0; r < 8; ++r)
      hbf[m0 + r + hi * 8][n0 + nl] = f2b(fmaxf(acc[i][r] + b2[n0 + nl], 0.f));
  }
  __syncthreads();

  // ---- GEMM3: [32,256] x [256,512] -> Hinit (fp32)
  for (int i = 0; i < 8; ++i) {
    int n0 = (nb * 8 + i) * 16;
    v8f c = gemm_tile(&hbf[0][0], 264, m0, Wibf, 256, n0, 256, lane);
    for (int r = 0; r < 8; ++r)
      Hinit[(long)(b0 + m0 + r + hi * 8) * 512 + n0 + nl] = c[r] + binit[n0 + nl];
  }
}

// ---------------------------------------------------------------------------
// GRU: 16 batch rows per workgroup, persistent over all timesteps.
// Gate buffer S (bf16): cols [0,768) = gi+gh+biases, cols [768,1024) = h_n+bhh_n
// Balanced tiling: every wave does 6 gate tiles + 2 h_n tiles.
// ---------------------------------------------------------------------------
__device__ __forceinline__ void cell_gemm(
    const bf16* __restrict__ A, int lda, int Kc, const bf16* __restrict__ Wc,
    const float* __restrict__ bS,
    const bf16* __restrict__ Ah, int ldah, const bf16* __restrict__ Wn,
    const float* __restrict__ bN,
    bf16 (*S)[1032], int lane, int w) {
  const int hi = (lane >> 4) & 1, nl = lane & 15;
  for (int i = 0; i < 6; ++i) {                       // gate tiles (48 total)
    int n0 = (w * 6 + i) * 16;
    v8f acc = gemm_tile(A, lda, 0, Wc, Kc, n0, Kc, lane);
    for (int r = 0; r < 8; ++r)
      S[r + hi * 8][n0 + nl] = f2b(acc[r] + bS[n0 + nl]);
  }
  for (int j = 0; j < 2; ++j) {                       // h_n tiles (16 total)
    int n0 = (w * 2 + j) * 16;
    v8f acc = gemm_tile(Ah, ldah, 0, Wn, 256, n0, 256, lane);
    for (int r = 0; r < 8; ++r)
      S[r + hi * 8][768 + n0 + nl] = f2b(acc[r] + bN[n0 + nl]);
  }
}

__global__ __launch_bounds__(256) void gru_kernel(
    const float* __restrict__ Hinit,
    const bf16* __restrict__ Wcat0, const bf16* __restrict__ Whn0,
    const float* __restrict__ bS0, const float* __restrict__ bN0,
    const bf16* __restrict__ Wcat1, const bf16* __restrict__ Whn1,
    const float* __restrict__ bS1, const float* __restrict__ bN1,
    const bf16* __restrict__ Wabf, const float* __restrict__ ba,
    float* __restrict__ out, const int* __restrict__ horizon, int B) {
  __shared__ bf16  S[16][1032];     // gate pre-activations (bf16)
  __shared__ bf16  a0buf[16][296];  // [a(8) | h0(256) | zero-pad to 288]
  __shared__ bf16  a1buf[16][520];  // [h0n(256) | h1(256)]
  __shared__ float lg[16][8];       // logits

  const int t = threadIdx.x, lane = t & 31, w = t >> 5;
  const int hi = (lane >> 4) & 1, nl = lane & 15;
  const int b0 = blockIdx.x * 16;
  const int T = horizon[0];
  const int row = t >> 4, cb = (t & 15) << 4;

  float h0r[16], h1r[16];
  {
    // torch .view(2,B,H) of [B,2H]: h0[b] = Hinit[b>>1, (b&1)*256+:],
    //                               h1[b] = Hinit[B/2 + (b>>1), (b&1)*256+:]
    int b = b0 + row;
    int half = (b & 1) * 256;
    long r0 = (long)(b >> 1) * 512 + half;
    long r1 = (long)((B >> 1) + (b >> 1)) * 512 + half;
    for (int i = 0; i < 16; ++i) {
      h0r[i] = Hinit[r0 + cb + i];
      h1r[i] = Hinit[r1 + cb + i];
      a0buf[row][8 + cb + i]   = f2b(h0r[i]);
      a1buf[row][256 + cb + i] = f2b(h1r[i]);
    }
  }
  if ((t & 15) == 0) for (int c = 0; c < 8; ++c)     a0buf[row][c] = f2b(0.f); // a0 = 0
  if ((t & 15) == 1) for (int c = 264; c < 296; ++c) a0buf[row][c] = f2b(0.f); // K pad
  __syncthreads();

  for (int st = 0; st < T; ++st) {
    // ---- GRU layer 0: A = [a | h0], K=288
    cell_gemm(&a0buf[0][0], 296, 288, Wcat0, bS0,
              &a0buf[0][8], 296, Whn0, bN0, S, lane, w);
    __syncthreads();
    for (int i = 0; i < 16; ++i) {
      int j = cb + i;
      float Sr = b2f(S[row][j]),       Sz = b2f(S[row][256 + j]);
      float Sn = b2f(S[row][512 + j]), hn = b2f(S[row][768 + j]);
      float rg = 1.f / (1.f + __expf(-Sr));
      float zg = 1.f / (1.f + __expf(-Sz));
      float ng = tanhf(Sn + (rg - 1.f) * hn);   // = tanh(i_n + r*h_n)
      float hv = (1.f - zg) * ng + zg * h0r[i];
      h0r[i] = hv;
      bf16 hb = f2b(hv);
      a0buf[row][8 + j] = hb;   // next step's layer-0 input h
      a1buf[row][j]     = hb;   // this step's layer-1 input x
    }
    __syncthreads();
    // ---- GRU layer 1: A = [h0n | h1], K=512
    cell_gemm(&a1buf[0][0], 520, 512, Wcat1, bS1,
              &a1buf[0][256], 520, Whn1, bN1, S, lane, w);
    __syncthreads();
    for (int i = 0; i < 16; ++i) {
      int j = cb + i;
      float Sr = b2f(S[row][j]),       Sz = b2f(S[row][256 + j]);
      float Sn = b2f(S[row][512 + j]), hn = b2f(S[row][768 + j]);
      float rg = 1.f / (1.f + __expf(-Sr));
      float zg = 1.f / (1.f + __expf(-Sz));
      float ng = tanhf(Sn + (rg - 1.f) * hn);
      float hv = (1.f - zg) * ng + zg * h1r[i];
      h1r[i] = hv;
      bf16 hb = f2b(hv);
      a1buf[row][j]       = hb; // h1n for the action head
      a1buf[row][256 + j] = hb; // next step's layer-1 hidden
    }
    __syncthreads();
    // ---- action head via WMMA: logits = h1n @ Wa^T + ba (wave 0, one tile)
    if (w == 0) {
      v8f c = gemm_tile(&a1buf[0][0], 520, 0, Wabf, 256, 0, 256, lane);
      if (nl < 8) {
        float bav = ba[nl];
        for (int r = 0; r < 8; ++r) {
          int M = r + hi * 8;
          float v = c[r] + bav;
          lg[M][nl] = v;
          out[((long)(b0 + M) * T + st) * 8 + nl] = v;
        }
      }
    }
    __syncthreads();
    // ---- softmax -> next action input
    if (t < 16) {
      float m = -1e30f;
      for (int c = 0; c < 8; ++c) m = fmaxf(m, lg[t][c]);
      float e[8], sum = 0.f;
      for (int c = 0; c < 8; ++c) { e[c] = __expf(lg[t][c] - m); sum += e[c]; }
      float inv = 1.f / sum;
      for (int c = 0; c < 8; ++c) a0buf[t][c] = f2b(e[c] * inv);
    }
    __syncthreads();
  }
}

// ---------------------------------------------------------------------------
extern "C" void kernel_launch(void* const* d_in, const int* in_sizes, int n_in,
                              void* d_out, int out_size, void* d_ws, size_t ws_size,
                              hipStream_t stream) {
  const float* ini   = (const float*)d_in[0];
  const float* fin   = (const float*)d_in[1];
  const int*   hor   = (const int*)  d_in[2];
  const float* W1    = (const float*)d_in[3];
  const float* b1    = (const float*)d_in[4];
  const float* lng   = (const float*)d_in[5];
  const float* lnb   = (const float*)d_in[6];
  const float* W2    = (const float*)d_in[7];
  const float* b2    = (const float*)d_in[8];
  const float* Wih0  = (const float*)d_in[9];
  const float* Whh0  = (const float*)d_in[10];
  const float* bih0  = (const float*)d_in[11];
  const float* bhh0  = (const float*)d_in[12];
  const float* Wih1  = (const float*)d_in[13];
  const float* Whh1  = (const float*)d_in[14];
  const float* bih1  = (const float*)d_in[15];
  const float* bhh1  = (const float*)d_in[16];
  const float* Wa    = (const float*)d_in[17];
  const float* ba    = (const float*)d_in[18];
  const float* Winit = (const float*)d_in[19];
  const float* binit = (const float*)d_in[20];
  float* out = (float*)d_out;

  const int B = in_sizes[0] / 512;

  // ---- workspace carve-up
  char* ws = (char*)d_ws;
  float* Hinit = (float*)ws;                       ws += (size_t)B * 512 * 4;
  bf16* W1bf   = (bf16*)ws;                        ws += 256 * 1024 * 2;
  bf16* W2bf   = (bf16*)ws;                        ws += 256 * 256 * 2;
  bf16* Wibf   = (bf16*)ws;                        ws += 512 * 256 * 2;
  bf16* Wcat0  = (bf16*)ws;                        ws += 768 * 288 * 2;
  bf16* Whn0   = (bf16*)ws;                        ws += 256 * 256 * 2;
  bf16* Wcat1  = (bf16*)ws;                        ws += 768 * 512 * 2;
  bf16* Whn1   = (bf16*)ws;                        ws += 256 * 256 * 2;
  bf16* Wabf   = (bf16*)ws;                        ws += 16 * 256 * 2;
  float* bS0   = (float*)ws;                       ws += 768 * 4;
  float* bN0   = (float*)ws;                       ws += 256 * 4;
  float* bS1   = (float*)ws;                       ws += 768 * 4;
  float* bN1   = (float*)ws;                       ws += 256 * 4;

  // ---- pack weights to bf16 (L2-resident)
  cvt_kernel<<<(256 * 1024 + 255) / 256, 256, 0, stream>>>(W1, W1bf, 256 * 1024);
  cvt_kernel<<<(256 * 256 + 255) / 256, 256, 0, stream>>>(W2, W2bf, 256 * 256);
  cvt_kernel<<<(512 * 256 + 255) / 256, 256, 0, stream>>>(Winit, Wibf, 512 * 256);
  pack_cat_kernel<<<(768 * 288 + 255) / 256, 256, 0, stream>>>(Wih0, Whh0, Wcat0, 8, 256, 288, 768 * 288);
  pack_cat_kernel<<<(768 * 512 + 255) / 256, 256, 0, stream>>>(Wih1, Whh1, Wcat1, 256, 256, 512, 768 * 512);
  pack_hn_kernel<<<(256 * 256 + 255) / 256, 256, 0, stream>>>(Whh0, Whn0);
  pack_hn_kernel<<<(256 * 256 + 255) / 256, 256, 0, stream>>>(Whh1, Whn1);
  pack_wa_kernel<<<16, 256, 0, stream>>>(Wa, Wabf);
  bias_kernel<<<3, 256, 0, stream>>>(bih0, bhh0, bS0, bN0);
  bias_kernel<<<3, 256, 0, stream>>>(bih1, bhh1, bS1, bN1);

  // ---- encoder MLP -> Hinit
  encoder_kernel<<<B / 32, 256, 0, stream>>>(ini, fin, W1bf, b1, lng, lnb,
                                             W2bf, b2, Wibf, binit, Hinit);
  // ---- persistent GRU + action head
  gru_kernel<<<B / 16, 256, 0, stream>>>(Hinit, Wcat0, Whn0, bS0, bN0,
                                         Wcat1, Whn1, bS1, bN1, Wabf, ba,
                                         out, hor, B);
}
